// ConvG_9440338116868
// MI455X (gfx1250) — matrix operand order repaired and checked by reading, not compiled, not run on hardware
//
#include <hip/hip_runtime.h>

// ---------------------------------------------------------------------------
// Gaussian-kernel conv (16,32,192,192) * (32,32,9,9) -> (16,32,192,192), pad 4
// Implicit-GEMM on v_wmma_f32_16x16x32_f16 (f16 inputs, f32 accumulation).
// Compute-bound (AI ~650 flop/byte; HBM floor ~6.5us) -> matrix pipe.
// Weights are generated on-device and stored directly in WMMA A-fragment
// lane order, so the conv inner loop fetches A with one v16h load per tile.
// ---------------------------------------------------------------------------

typedef __attribute__((ext_vector_type(16))) _Float16 v16h;
typedef __attribute__((ext_vector_type(8)))  float    v8f;

#define H      192
#define W      192
#define CIN    32
#define COUT   32
#define KS     9
#define PADR   4
#define TIL    16                 // output tile edge (pixels)
#define LR     (TIL + KS - 1)     // 24: staged rows/cols incl. halo
#define CI_PAD 40                 // pad 32->40 f16 (80 B): breaks LDS bank-conflict
                                  // periodicity, keeps 16 B alignment for v16h
#define TAPFRAG (2 * 32 * 16)     // f16 per tap: 2 co-tiles x 32 lanes x 16 elems

// ---------------------------------------------------------------------------
// Kernel 1: build normalized anisotropic Gaussian 9x9 kernels, emit f16
// weights pre-swizzled into WMMA 16x16x32 A-fragment order:
//   wbuf[((kk*2 + t)*32 + lane)*16 + e]
// where t = co/16, lane = (co%15..) = (co&15) + 16*g, and (g,e) encode ci per
// the contiguous k order shared with B: K(e,g) = e + 16g.
// ---------------------------------------------------------------------------
__global__ __launch_bounds__(256)
void build_kernels(const float* __restrict__ wparam, _Float16* __restrict__ wbuf) {
  int idx = blockIdx.x * blockDim.x + threadIdx.x;   // 0..1023
  int co = idx >> 5, ci = idx & 31;
  const float* wp = wparam + (size_t)(co * CIN + ci) * 3;
  float sx = wp[0], sy = wp[1], th = wp[2];
  float rad = th * 0.017453292519943295f;            // reference: th/180*pi
  float c = cosf(rad), s = sinf(rad);
  float sx2 = sx * sx, sy2 = sy * sy;
  float a = c * c * sx2 + s * s * sy2;
  float b = c * s * (sx2 - sy2);
  float d = s * s * sx2 + c * c * sy2;
  float det = a * d - b * b;
  float ia = d / det, ib = -b / det, idm = a / det;

  float sum = 0.f;
  for (int kh = 0; kh < KS; ++kh) {
    float yy = (float)(kh - KS / 2);
    for (int kw = 0; kw < KS; ++kw) {
      float xx = (float)(kw - KS / 2);
      float q = ia * xx * xx + 2.f * ib * xx * yy + idm * yy * yy;
      sum += expf(-0.5f * q);
    }
  }
  float inv = 1.f / sum;

  // Destination swizzle for this (co, ci), constant across taps.
  int t    = co >> 4;
  int m    = co & 15;
  int g    = ci >> 4;
  int e    = ci & 15;
  int lane = m + 16 * g;

  for (int kh = 0; kh < KS; ++kh) {
    float yy = (float)(kh - KS / 2);
    for (int kw = 0; kw < KS; ++kw) {
      float xx = (float)(kw - KS / 2);
      float q = ia * xx * xx + 2.f * ib * xx * yy + idm * yy * yy;
      float v = expf(-0.5f * q) * inv;
      int kk = kh * KS + kw;
      wbuf[(size_t)kk * TAPFRAG + ((t * 32 + lane) << 4) + e] = (_Float16)v;
    }
  }
}

// ---------------------------------------------------------------------------
// Kernel 2: WMMA implicit-GEMM conv.
// Block = 512 threads = 16 wave32s. Block tile: 16 rows x 16 cols x 32 co,
// one batch image. Wave w computes output row h0+w for all 32 co:
// two v8f accumulators (co 0..15, 16..31) sharing each B fragment.
// Per (kh,kw) tap: one K=32 WMMA step over all input channels.
// ---------------------------------------------------------------------------
__global__ __launch_bounds__(512)
void conv_wmma(const float* __restrict__ x, const _Float16* __restrict__ wbuf,
               const float* __restrict__ bias, float* __restrict__ out) {
  __shared__ __align__(16) _Float16 tile[LR * LR * CI_PAD];   // 46,080 B

  const int w0  = blockIdx.x * TIL;
  const int h0  = blockIdx.y * TIL;
  const int b   = blockIdx.z;
  const int tid = threadIdx.x;

  // Cooperative halo fill: 32ci x 24x24 = 18432 f16, 36 per thread.
  // Consecutive lanes read consecutive gx -> coalesced global loads.
  for (int j = 0; j < (CIN * LR * LR) / 512; ++j) {
    int i   = tid + 512 * j;
    int ci  = i / (LR * LR);
    int rem = i - ci * (LR * LR);
    int r   = rem / LR;
    int c   = rem - r * LR;
    int gy  = h0 - PADR + r;
    int gx  = w0 - PADR + c;
    float v = 0.f;
    if (gy >= 0 && gy < H && gx >= 0 && gx < W)
      v = x[(((size_t)b * CIN + ci) * H + gy) * W + gx];
    tile[(r * LR + c) * CI_PAD + ci] = (_Float16)v;
  }
  __syncthreads();

  const int wave = tid >> 5;       // 0..15 -> output row inside tile
  const int lane = tid & 31;
  const int n    = lane & 15;      // B column (pixel)
  const int g    = lane >> 4;      // lane half selects the ci sub-range of B

  v8f acc0 = {0.f, 0.f, 0.f, 0.f, 0.f, 0.f, 0.f, 0.f};
  v8f acc1 = acc0;

  for (int kh = 0; kh < KS; ++kh) {
    // B fragment source row in LDS for this tap row
    const _Float16* brow = &tile[((wave + kh) * LR) * CI_PAD + 16 * g];
    for (int kw = 0; kw < KS; ++kw) {
      const int kk = kh * KS + kw;
      const _Float16* ap = wbuf + (size_t)kk * TAPFRAG + (lane << 4);
      if (kk < KS * KS - 1)
        __builtin_prefetch(ap + TAPFRAG, 0, 1);      // global_prefetch_b8

      // B (32x16): lane holds ci = 16g+e for pixel n -> one aligned v16h load
      v16h bfrag = *(const v16h*)(brow + (n + kw) * CI_PAD);

      // A fragments pre-swizzled: one aligned 32 B load per co-tile
      v16h af0 = *(const v16h*)(ap);                 // co 0..15
      v16h af1 = *(const v16h*)(ap + 32 * 16);       // co 16..31

      acc0 = __builtin_amdgcn_wmma_f32_16x16x32_f16(false, af0, false, bfrag,
                                                    (short)0, acc0, false, false);
      acc1 = __builtin_amdgcn_wmma_f32_16x16x32_f16(false, af1, false, bfrag,
                                                    (short)0, acc1, false, false);
    }
  }

  // C/D layout: VGPR r, lane l -> M = r + 8*(l/16), N = l%16
  const int row = h0 + wave;
  const int col = w0 + n;
#pragma unroll
  for (int r = 0; r < 8; ++r) {
    int m   = r + 8 * g;
    int coA = m;
    int coB = m + 16;
    out[(((size_t)b * COUT + coA) * H + row) * W + col] = acc0[r] + bias[coA];
    out[(((size_t)b * COUT + coB) * H + row) * W + col] = acc1[r] + bias[coB];
  }
}

// ---------------------------------------------------------------------------
// Launch: inputs are (x f32, weight f32 (32,32,3), bias f32 (32,)),
// output f32 (16,32,192,192). d_ws holds the f16 fragment bank (165,888 B).
// ---------------------------------------------------------------------------
extern "C" void kernel_launch(void* const* d_in, const int* in_sizes, int n_in,
                              void* d_out, int out_size, void* d_ws, size_t ws_size,
                              hipStream_t stream) {
  (void)in_sizes; (void)n_in; (void)out_size; (void)ws_size;
  const float* x      = (const float*)d_in[0];
  const float* wparam = (const float*)d_in[1];
  const float* bias   = (const float*)d_in[2];
  float*       out    = (float*)d_out;
  _Float16*    wbuf   = (_Float16*)d_ws;    // 81 taps * 1024 f16 = 165,888 bytes

  build_kernels<<<dim3(4), dim3(256), 0, stream>>>(wparam, wbuf);
  conv_wmma<<<dim3(W / TIL, H / TIL, 16), dim3(512), 0, stream>>>(x, wbuf, bias, out);
}
